// GroupLinear_60945585930502
// MI455X (gfx1250) — matrix operand order repaired and checked by reading, compiled, not run on hardware
//
#include <hip/hip_runtime.h>

#define L_DIM   2048
#define O_DIM   1024
#define IN_DIM  1024
#define BS      64
#define G_DIM   16
#define K_DIM   65536     // G_DIM * BS * BS

#define MT      64        // M tile per workgroup
#define NT      256       // N tile per workgroup
#define PITCH   68        // LDS row pitch in floats (16B-aligned rows, bank stride 4)

typedef __attribute__((ext_vector_type(16))) __bf16 v16bf;
typedef __attribute__((ext_vector_type(8)))  float  v8f;

// ---------------------------------------------------------------------------
// Kernel 1: convert W fp32 [O, K] -> bf16 [O, K] in workspace (one pass).
// ---------------------------------------------------------------------------
__global__ void __launch_bounds__(256)
wconv_kernel(const float4* __restrict__ w, uint4* __restrict__ wb)
{
    size_t i = (size_t)blockIdx.x * 256 + threadIdx.x;   // one uint4 (8 bf16) out
    float4 a = w[2 * i];
    float4 b = w[2 * i + 1];
    union { __bf16 h[8]; uint4 u; } p;
    p.h[0] = (__bf16)a.x; p.h[1] = (__bf16)a.y; p.h[2] = (__bf16)a.z; p.h[3] = (__bf16)a.w;
    p.h[4] = (__bf16)b.x; p.h[5] = (__bf16)b.y; p.h[6] = (__bf16)b.z; p.h[7] = (__bf16)b.w;
    wb[i] = p.u;
}

// ---------------------------------------------------------------------------
// Kernel 2: out[l,o] = sum_{g,i,j} x1[l,gi]*x2[l,gj]*W[o,gij] + b[o]
// GEMM  M=2048 N=1024 K=65536, A built on the fly, B = bf16 W from workspace.
// ---------------------------------------------------------------------------
__global__ void __launch_bounds__(256)
bilinear_gemm_kernel(const float* __restrict__ x1,
                     const float* __restrict__ x2,
                     const unsigned short* __restrict__ wbu,  // bf16 bits [O,K]
                     const float* __restrict__ bias,
                     float* __restrict__ out)
{
    __shared__ float lds1[MT * PITCH];   // x1 block: [m][j]  (column i read per step)
    __shared__ float lds2[MT * PITCH];   // x2 block: [m][j]

    const int tid  = threadIdx.x;
    const int lane = tid & 31;
    const int hlf  = lane >> 4;          // 0: lanes 0-15, 1: lanes 16-31
    const int l16  = lane & 15;
    const int wave = tid >> 5;
    const int wr   = wave & 1;           // M sub-block: rows [32*wr, 32*wr+32)
    const int wc   = wave >> 1;          // N strip:     cols [64*wc, 64*wc+64)

    const int m0 = blockIdx.y * MT;
    const int n0 = blockIdx.x * NT;

    v8f acc[2][4];
    const v8f vzero = {0.f, 0.f, 0.f, 0.f, 0.f, 0.f, 0.f, 0.f};
#pragma unroll
    for (int a = 0; a < 2; ++a)
#pragma unroll
        for (int n = 0; n < 4; ++n) acc[a][n] = vzero;

    float bv[4];
#pragma unroll
    for (int nst = 0; nst < 4; ++nst)
        bv[nst] = bias[n0 + wc * 64 + nst * 16 + l16];

#pragma unroll 1
    for (int g = 0; g < G_DIM; ++g) {
        __syncthreads();
        // Cooperatively stage x1/x2 64x64 fp32 blocks for this g.
#pragma unroll
        for (int it = 0; it < 4; ++it) {
            int idx = tid + it * 256;            // 0..1023 float4 slots
            int row = idx >> 4;                  // 0..63
            int c4  = idx & 15;                  // float4 column
            const size_t goff = (size_t)(m0 + row) * IN_DIM + g * BS + c4 * 4;
            *(float4*)(lds1 + row * PITCH + c4 * 4) = *(const float4*)(x1 + goff);
            *(float4*)(lds2 + row * PITCH + c4 * 4) = *(const float4*)(x2 + goff);
        }
        __syncthreads();

#pragma unroll 1
        for (int i = 0; i < BS; ++i) {
            // scalar x1[m, g*64+i] for this lane's two output rows
            float x1v[2];
            x1v[0] = lds1[(wr * 32 +  0 + l16) * PITCH + i];
            x1v[1] = lds1[(wr * 32 + 16 + l16) * PITCH + i];

#pragma unroll
            for (int jb = 0; jb < 2; ++jb) {     // two K=32 steps inside the j-block
                // B fragments: two runs of 8 bf16 per lane (K 8h..8h+7, 16+8h..), straight from global.
                union BU { int4 q[2]; v16bf v; } bfr[4];
                const size_t kbase =
                    (size_t)g * 4096 + (size_t)i * 64 + jb * 32 + hlf * 8;
#pragma unroll
                for (int nst = 0; nst < 4; ++nst) {
                    const unsigned short* bp =
                        wbu + (size_t)(n0 + wc * 64 + nst * 16 + l16) * K_DIM + kbase;
                    bfr[nst].q[0] = *(const int4*)(bp);
                    bfr[nst].q[1] = *(const int4*)(bp + 16);
                }

#pragma unroll
                for (int msub = 0; msub < 2; ++msub) {
                    // A fragment layout (16-bit A 16x32): lanes 0-15 hold K 0-7,16-23;
                    // lanes 16-31 hold K 8-15,24-31 -> j offsets jb*32 + hlf*8 (+16).
                    const float* xr =
                        lds2 + (wr * 32 + msub * 16 + l16) * PITCH + jb * 32 + hlf * 8;
                    float4 a0 = *(const float4*)(xr);
                    float4 a1 = *(const float4*)(xr + 4);
                    float4 a2 = *(const float4*)(xr + 16);
                    float4 a3 = *(const float4*)(xr + 20);
                    const float s = x1v[msub];
                    v16bf af;
                    af[0]  = (__bf16)(s * a0.x); af[1]  = (__bf16)(s * a0.y);
                    af[2]  = (__bf16)(s * a0.z); af[3]  = (__bf16)(s * a0.w);
                    af[4]  = (__bf16)(s * a1.x); af[5]  = (__bf16)(s * a1.y);
                    af[6]  = (__bf16)(s * a1.z); af[7]  = (__bf16)(s * a1.w);
                    af[8]  = (__bf16)(s * a2.x); af[9]  = (__bf16)(s * a2.y);
                    af[10] = (__bf16)(s * a2.z); af[11] = (__bf16)(s * a2.w);
                    af[12] = (__bf16)(s * a3.x); af[13] = (__bf16)(s * a3.y);
                    af[14] = (__bf16)(s * a3.z); af[15] = (__bf16)(s * a3.w);

#pragma unroll
                    for (int nst = 0; nst < 4; ++nst)
                        acc[msub][nst] = __builtin_amdgcn_wmma_f32_16x16x32_bf16(
                            false, af, false, bfr[nst].v, (short)0,
                            acc[msub][nst], false, false);
                }
            }
        }
    }

    // Epilogue: C/D layout -> VGPR v holds M = v + 8*hlf, column N = l16.
#pragma unroll
    for (int msub = 0; msub < 2; ++msub)
#pragma unroll
        for (int nst = 0; nst < 4; ++nst)
#pragma unroll
            for (int v = 0; v < 8; ++v) {
                int m = m0 + wr * 32 + msub * 16 + 8 * hlf + v;
                int n = n0 + wc * 64 + nst * 16 + l16;
                out[(size_t)m * O_DIM + n] = acc[msub][nst][v] + bv[nst];
            }
}

// ---------------------------------------------------------------------------
extern "C" void kernel_launch(void* const* d_in, const int* in_sizes, int n_in,
                              void* d_out, int out_size, void* d_ws, size_t ws_size,
                              hipStream_t stream)
{
    const float* x1 = (const float*)d_in[0];   // [2048, 1024]
    const float* x2 = (const float*)d_in[1];   // [2048, 1024]
    const float* W  = (const float*)d_in[2];   // [1024, 65536]
    const float* b  = (const float*)d_in[3];   // [1024]
    float* out = (float*)d_out;                // [2048, 1024]

    // bf16 copy of W lives in workspace: 1024*65536*2 = 128 MiB (fits global L2).
    unsigned short* wb = (unsigned short*)d_ws;

    const size_t welems = (size_t)O_DIM * K_DIM;          // 2^26
    wconv_kernel<<<(unsigned)(welems / (256 * 8)), 256, 0, stream>>>(
        (const float4*)W, (uint4*)wb);

    dim3 grid(O_DIM / NT, L_DIM / MT);                    // (4, 32)
    bilinear_gemm_kernel<<<grid, 256, 0, stream>>>(x1, x2, wb, b, out);
}
